// Inter_Patch_MoE_30966714204612
// MI455X (gfx1250) — matrix operand order repaired and checked by reading, not compiled, not run on hardware
//
#include <hip/hip_runtime.h>
#include <math.h>

typedef __attribute__((ext_vector_type(16))) __bf16 bf16x16;
typedef __attribute__((ext_vector_type(8)))  __bf16 bf16x8;
typedef __attribute__((ext_vector_type(8)))  float  floatx8;

#define BM 128
#define BN 128
#define BK 32
#define LDS_STRIDE 40   // bf16 elements; 80 bytes = 5*16: 16B-aligned rows, skewed banks

__device__ __forceinline__ float gelu_exact(float v) {
    return 0.5f * v * (1.0f + erff(v * 0.70710678118654752f));
}

// ---------------------------------------------------------------------------
// GEMM: C[M x N] = act(A[M x K] @ W[K x N] + bias[N]); fp32 in memory,
// bf16 WMMA (fp32 accumulate). K must be a multiple of 32.
// 128x128 block tile, 8 waves, each wave computes 32x64 (8 WMMA per k-step).
// LDS double-buffered; next tile's global loads overlap current WMMAs.
// ---------------------------------------------------------------------------
__global__ __launch_bounds__(256) void wmma_gemm_kernel(
    const float* __restrict__ A, const float* __restrict__ W,
    const float* __restrict__ bias, float* __restrict__ C,
    int M, int N, int K, int act)
{
    __shared__ __attribute__((aligned(16))) __bf16 As[2][BM][LDS_STRIDE];
    __shared__ __attribute__((aligned(16))) __bf16 Bs[2][BN][LDS_STRIDE];

    const int tid  = threadIdx.x;
    const int wave = tid >> 5;
    const int lane = tid & 31;
    const int wm   = wave >> 1;      // 0..3  (32-row group)
    const int wn   = wave & 1;       // 0..1  (64-col group)
    const int hlf  = lane >> 4;
    const int l16  = lane & 15;

    const int m0 = blockIdx.y * BM;
    const int n0 = blockIdx.x * BN;

    // Staging assignment (256 threads):
    //   A: row = tid>>1 (0..127), 16 contiguous floats at seg=(tid&1)*16
    //   W: col = tid&127, 16 k values starting at kseg=(tid>>7)*16
    const int arow = tid >> 1;
    const int aseg = (tid & 1) * 16;
    const int bcol = tid & 127;
    const int kseg = (tid >> 7) * 16;
    const int gm_stage = m0 + arow;
    const int gn_stage = n0 + bcol;

    // Accumulators: 2 (row groups) x 4 (col groups) 16x16 f32 tiles.
    floatx8 acc[2][4];
    #pragma unroll
    for (int t = 0; t < 4; ++t) {
        int n = n0 + wn * 64 + t * 16 + l16;
        float bv = (n < N) ? bias[n] : 0.0f;
        #pragma unroll
        for (int r = 0; r < 8; ++r) { acc[0][t][r] = bv; acc[1][t][r] = bv; }
    }

    float ar[16], wr[16];

    // ---- tile loader: global -> registers (vectorized, uniform guards) ----
    auto load_tile = [&](int k0) {
        if (gm_stage < M) {
            const float* src = A + (size_t)gm_stage * K + k0 + aseg;
            #pragma unroll
            for (int i = 0; i < 16; i += 4) {
                float4 v = *(const float4*)(src + i);
                ar[i] = v.x; ar[i+1] = v.y; ar[i+2] = v.z; ar[i+3] = v.w;
            }
        } else {
            #pragma unroll
            for (int i = 0; i < 16; ++i) ar[i] = 0.0f;
        }
        if (gn_stage < N) {
            const float* src = W + (size_t)(k0 + kseg) * N + gn_stage;
            #pragma unroll
            for (int i = 0; i < 16; ++i) wr[i] = src[(size_t)i * N];
        } else {
            #pragma unroll
            for (int i = 0; i < 16; ++i) wr[i] = 0.0f;
        }
    };
    // ---- registers -> LDS (bf16, 16B vector stores) ----
    auto store_tile = [&](int buf) {
        bf16x8 p0, p1;
        #pragma unroll
        for (int i = 0; i < 8; ++i) { p0[i] = (__bf16)ar[i]; p1[i] = (__bf16)ar[8 + i]; }
        *(bf16x8*)&As[buf][arow][aseg]     = p0;
        *(bf16x8*)&As[buf][arow][aseg + 8] = p1;
        #pragma unroll
        for (int i = 0; i < 8; ++i) { p0[i] = (__bf16)wr[i]; p1[i] = (__bf16)wr[8 + i]; }
        *(bf16x8*)&Bs[buf][bcol][kseg]     = p0;
        *(bf16x8*)&Bs[buf][bcol][kseg + 8] = p1;
    };

    load_tile(0);
    store_tile(0);
    __syncthreads();

    const int ktiles = K / BK;
    for (int kt = 0; kt < ktiles; ++kt) {
        const int cur = kt & 1;
        if (kt + 1 < ktiles) load_tile((kt + 1) * BK);   // overlap with WMMAs below

        // A fragments (16x32 bf16, ISA lane layout): lane row = l16,
        // elems 0..7 -> K = 8*hlf.., elems 8..15 -> K = 16+8*hlf..
        union { bf16x16 v; bf16x8 h[2]; } af[2];
        #pragma unroll
        for (int am = 0; am < 2; ++am) {
            const __bf16* ap = &As[cur][wm * 32 + am * 16 + l16][0];
            af[am].h[0] = *(const bf16x8*)(ap + 8 * hlf);
            af[am].h[1] = *(const bf16x8*)(ap + 16 + 8 * hlf);
        }
        // B fragments (32x16 bf16): lane col = l16, elems 0..7 -> K = 8*hlf.., elems 8..15 -> K = 16+8*hlf..
        #pragma unroll
        for (int t = 0; t < 4; ++t) {
            union { bf16x16 v; bf16x8 h[2]; } bfr;
            const __bf16* bp = &Bs[cur][wn * 64 + t * 16 + l16][0];
            bfr.h[0] = *(const bf16x8*)(bp + 8 * hlf);
            bfr.h[1] = *(const bf16x8*)(bp + 16 + 8 * hlf);
            acc[0][t] = __builtin_amdgcn_wmma_f32_16x16x32_bf16(
                false, af[0].v, false, bfr.v, (short)0, acc[0][t], false, false);
            acc[1][t] = __builtin_amdgcn_wmma_f32_16x16x32_bf16(
                false, af[1].v, false, bfr.v, (short)0, acc[1][t], false, false);
        }

        if (kt + 1 < ktiles) store_tile(cur ^ 1);
        __syncthreads();
    }

    // ---- epilogue: activation + store ----
    #pragma unroll
    for (int am = 0; am < 2; ++am) {
        #pragma unroll
        for (int t = 0; t < 4; ++t) {
            int n = n0 + wn * 64 + t * 16 + l16;
            if (n < N) {
                #pragma unroll
                for (int r = 0; r < 8; ++r) {
                    int m = m0 + wm * 32 + am * 16 + hlf * 8 + r;
                    if (m < M) {
                        float v = acc[am][t][r];
                        if (act == 1) v = gelu_exact(v);
                        C[(size_t)m * N + n] = v;
                    }
                }
            }
        }
    }
}

// ---------------------------------------------------------------------------
// Multi-scale gate features: gi[b] = [mean | min4,stdm4,max4 | ps=8 | ps=16]
// ---------------------------------------------------------------------------
__global__ void scale_features_kernel(const float* __restrict__ x,
                                      float* __restrict__ gi,
                                      int B, int T, int D)
{
    int idx = blockIdx.x * blockDim.x + threadIdx.x;
    if (idx >= B * D) return;
    int b = idx / D, d = idx % D;
    const float* xp = x + (size_t)b * T * D + d;

    const int psm[3] = {3, 7, 15};
    const float psf[3] = {4.f, 8.f, 16.f};
    float tot = 0.f;
    float s[3] = {0.f, 0.f, 0.f}, ss[3] = {0.f, 0.f, 0.f};
    float mn[3], mx[3], gmn[3], gmx[3], stds[3];
    #pragma unroll
    for (int j = 0; j < 3; ++j) {
        mn[j] = 3.0e38f; mx[j] = -3.0e38f;
        gmn[j] = 3.0e38f; gmx[j] = -3.0e38f; stds[j] = 0.f;
    }

    for (int t = 0; t < T; ++t) {
        float v = xp[(size_t)t * D];
        tot += v;
        #pragma unroll
        for (int j = 0; j < 3; ++j) {
            s[j] += v; ss[j] += v * v;
            mn[j] = fminf(mn[j], v); mx[j] = fmaxf(mx[j], v);
            if (((t + 1) & psm[j]) == 0) {
                float n = psf[j];
                float var = (ss[j] - s[j] * s[j] / n) / (n - 1.f);
                stds[j] += sqrtf(fmaxf(var, 0.f));
                gmn[j] = fminf(gmn[j], mn[j]);
                gmx[j] = fmaxf(gmx[j], mx[j]);
                s[j] = 0.f; ss[j] = 0.f; mn[j] = 3.0e38f; mx[j] = -3.0e38f;
            }
        }
    }
    float* g = gi + (size_t)b * (10 * D);
    g[d] = tot / (float)T;
    #pragma unroll
    for (int j = 0; j < 3; ++j) {
        int base = D + j * 3 * D;
        float npatch = (float)T / psf[j];
        g[base + d]         = gmn[j];
        g[base + D + d]     = stds[j] / npatch;
        g[base + 2 * D + d] = gmx[j];
    }
}

// ---------------------------------------------------------------------------
// Patch means (ps=4,8,16) + context mean, single pass.
// ---------------------------------------------------------------------------
__global__ void patch_means_kernel(const float* __restrict__ x,
                                   float* __restrict__ m4, float* __restrict__ m8,
                                   float* __restrict__ m16, float* __restrict__ ctx,
                                   int B, int T, int D)
{
    int idx = blockIdx.x * blockDim.x + threadIdx.x;
    if (idx >= B * D) return;
    int b = idx / D, d = idx % D;
    const float* xp = x + (size_t)b * T * D + d;
    int NP4 = T / 4;
    float tot = 0.f, a8 = 0.f, a16 = 0.f;
    for (int p = 0; p < NP4; ++p) {
        float s4 = 0.f;
        #pragma unroll
        for (int q = 0; q < 4; ++q) s4 += xp[(size_t)(p * 4 + q) * D];
        m4[((size_t)b * NP4 + p) * D + d] = s4 * 0.25f;
        a8 += s4; a16 += s4; tot += s4;
        if (p & 1)        { m8 [((size_t)b * (NP4/2) + (p >> 1)) * D + d] = a8  * 0.125f;  a8  = 0.f; }
        if ((p & 3) == 3) { m16[((size_t)b * (NP4/4) + (p >> 2)) * D + d] = a16 * 0.0625f; a16 = 0.f; }
    }
    ctx[(size_t)b * D + d] = tot / (float)T;
}

// ---------------------------------------------------------------------------
// logits = h @ gate_w2 + b2 ; top-2 softmax -> coeff[b][3]
// ---------------------------------------------------------------------------
__global__ void gate_topk_kernel(const float* __restrict__ h,
                                 const float* __restrict__ w2,
                                 const float* __restrict__ b2,
                                 float* __restrict__ coeff, int F)
{
    int b = blockIdx.x;
    __shared__ float red[256];
    __shared__ float logits[3];
    float part[3] = {0.f, 0.f, 0.f};
    for (int i = threadIdx.x; i < F; i += 256) {
        float hv = h[(size_t)b * F + i];
        part[0] += hv * w2[i * 3 + 0];
        part[1] += hv * w2[i * 3 + 1];
        part[2] += hv * w2[i * 3 + 2];
    }
    for (int e = 0; e < 3; ++e) {
        red[threadIdx.x] = part[e];
        __syncthreads();
        for (int s = 128; s > 0; s >>= 1) {
            if (threadIdx.x < s) red[threadIdx.x] += red[threadIdx.x + s];
            __syncthreads();
        }
        if (threadIdx.x == 0) logits[e] = red[0] + b2[e];
        __syncthreads();
    }
    if (threadIdx.x == 0) {
        int i1 = 0; float v1 = logits[0];
        if (logits[1] > v1) { v1 = logits[1]; i1 = 1; }
        if (logits[2] > v1) { v1 = logits[2]; i1 = 2; }
        int i2 = -1; float v2 = -3.0e38f;
        for (int e = 0; e < 3; ++e)
            if (e != i1 && logits[e] > v2) { v2 = logits[e]; i2 = e; }
        float e2 = expf(v2 - v1);
        float inv = 1.0f / (1.0f + e2);
        for (int e = 0; e < 3; ++e) {
            float c = (e == i1) ? inv : ((e == i2) ? e2 * inv : 0.0f);
            coeff[b * 3 + e] = c;
        }
    }
}

// ---------------------------------------------------------------------------
// Per-expert context gate: w[b] = sigmoid(gelu(ctx@gw1+gb1) @ gw2 + gb2)
// ---------------------------------------------------------------------------
__global__ void ctx_gate_kernel(const float* __restrict__ ctx,
                                const float* __restrict__ gw1, const float* __restrict__ gb1,
                                const float* __restrict__ gw2, const float* __restrict__ gb2,
                                float* __restrict__ wout, int D)
{
    int b = blockIdx.x;
    __shared__ float red[256];
    const float* c = ctx + (size_t)b * D;
    float part = 0.f;
    for (int j = threadIdx.x; j < D; j += 256) {
        float a = 0.f;
        for (int k = 0; k < D; ++k) a += c[k] * gw1[(size_t)k * D + j];
        a = gelu_exact(a + gb1[j]);
        part += a * gw2[j];
    }
    red[threadIdx.x] = part;
    __syncthreads();
    for (int s = 128; s > 0; s >>= 1) {
        if (threadIdx.x < s) red[threadIdx.x] += red[threadIdx.x + s];
        __syncthreads();
    }
    if (threadIdx.x == 0)
        wout[b] = 1.0f / (1.0f + expf(-(red[0] + gb2[0])));
}

__global__ void zero_kernel(float* __restrict__ out, int n) {
    int i = blockIdx.x * blockDim.x + threadIdx.x;
    if (i < n) out[i] = 0.0f;
}

// ---------------------------------------------------------------------------
// out += coeff[b,e] * ( w[b]*intra[i] + (1-w[b])*inter[bn,d] + x[i] )
// (intra flat index == x flat index since flat = x.reshape)
// ---------------------------------------------------------------------------
__global__ void fuse_acc_kernel(const float* __restrict__ x,
                                const float* __restrict__ intra,
                                const float* __restrict__ inter,
                                const float* __restrict__ wg,
                                const float* __restrict__ coeff,
                                int e, int ps, float* __restrict__ out,
                                int B, int T, int D)
{
    int i = blockIdx.x * blockDim.x + threadIdx.x;
    if (i >= B * T * D) return;
    int d = i % D;
    int bt = i / D;
    int t = bt % T;
    int b = bt / T;
    float c = coeff[b * 3 + e];
    float w = wg[b];
    int n = t / ps;
    int N = T / ps;
    float itv = intra[(size_t)i];
    float inv = inter[((size_t)b * N + n) * D + d];
    float fused = w * itv + (1.0f - w) * inv;
    out[(size_t)i] += c * (fused + x[(size_t)i]);
}

// ---------------------------------------------------------------------------
extern "C" void kernel_launch(void* const* d_in, const int* in_sizes, int n_in,
                              void* d_out, int out_size, void* d_ws, size_t ws_size,
                              hipStream_t stream) {
    (void)in_sizes; (void)n_in; (void)out_size; (void)ws_size;
    const int B = 8, T = 512, D = 512, F = 2048;
    const int psizes[3] = {4, 8, 16};

    const float* x = (const float*)d_in[0];
    // expert e params at 1 + e*12: iw1 ib1 iw2 ib2 ew1 eb1 ew2 eb2 gw1 gb1 gw2 gb2
    const float* gate_w1 = (const float*)d_in[37];
    const float* gate_b1 = (const float*)d_in[38];
    const float* gate_w2 = (const float*)d_in[39];
    const float* gate_b2 = (const float*)d_in[40];
    float* out = (float*)d_out;

    char* ws = (char*)d_ws;
    size_t off = 0;
    auto carve = [&](size_t bytes) -> float* {
        float* p = (float*)(ws + off);
        off = (off + bytes + 255) & ~(size_t)255;
        return p;
    };
    float* GI    = carve((size_t)B * 10 * D * 4);
    float* H     = carve((size_t)B * F * 4);
    float* COEFF = carve((size_t)B * 3 * 4);
    float* M4    = carve((size_t)B * (T/4)  * D * 4);
    float* M8    = carve((size_t)B * (T/8)  * D * 4);
    float* M16   = carve((size_t)B * (T/16) * D * 4);
    float* CTX   = carve((size_t)B * D * 4);
    float* WG    = carve((size_t)3 * B * 4);
    float* MID   = carve((size_t)(B*T/4) * F * 4);
    float* INTRA = carve((size_t)B * T * D * 4);
    float* IMID  = carve((size_t)(B*T/4) * F * 4);
    float* INTER = carve((size_t)(B*T/4) * D * 4);

    scale_features_kernel<<<(B * D + 255) / 256, 256, 0, stream>>>(x, GI, B, T, D);
    {
        dim3 g((F + BN - 1) / BN, (B + BM - 1) / BM);
        wmma_gemm_kernel<<<g, 256, 0, stream>>>(GI, gate_w1, gate_b1, H, B, F, 10 * D, 1);
    }
    gate_topk_kernel<<<B, 256, 0, stream>>>(H, gate_w2, gate_b2, COEFF, F);
    patch_means_kernel<<<(B * D + 255) / 256, 256, 0, stream>>>(x, M4, M8, M16, CTX, B, T, D);
    zero_kernel<<<(B * T * D + 255) / 256, 256, 0, stream>>>(out, B * T * D);

    for (int e = 0; e < 3; ++e) {
        int ps = psizes[e];
        int Me = B * T / ps;           // 1024 / 512 / 256
        int KD = ps * D;               // 2048 / 4096 / 8192
        const float* iw1 = (const float*)d_in[1 + e * 12 + 0];
        const float* ib1 = (const float*)d_in[1 + e * 12 + 1];
        const float* iw2 = (const float*)d_in[1 + e * 12 + 2];
        const float* ib2 = (const float*)d_in[1 + e * 12 + 3];
        const float* ew1 = (const float*)d_in[1 + e * 12 + 4];
        const float* eb1 = (const float*)d_in[1 + e * 12 + 5];
        const float* ew2 = (const float*)d_in[1 + e * 12 + 6];
        const float* eb2 = (const float*)d_in[1 + e * 12 + 7];
        const float* gw1 = (const float*)d_in[1 + e * 12 + 8];
        const float* gb1 = (const float*)d_in[1 + e * 12 + 9];
        const float* gw2 = (const float*)d_in[1 + e * 12 + 10];
        const float* gb2 = (const float*)d_in[1 + e * 12 + 11];
        const float* inter_in = (e == 0) ? M4 : (e == 1) ? M8 : M16;

        {
            dim3 g1((F + BN - 1) / BN, (Me + BM - 1) / BM);
            wmma_gemm_kernel<<<g1, 256, 0, stream>>>(x, iw1, ib1, MID, Me, F, KD, 0);
            dim3 g2((KD + BN - 1) / BN, (Me + BM - 1) / BM);
            wmma_gemm_kernel<<<g2, 256, 0, stream>>>(MID, iw2, ib2, INTRA, Me, KD, F, 0);
        }
        {
            dim3 g3((F + BN - 1) / BN, (Me + BM - 1) / BM);
            wmma_gemm_kernel<<<g3, 256, 0, stream>>>(inter_in, ew1, eb1, IMID, Me, F, D, 0);
            dim3 g4((D + BN - 1) / BN, (Me + BM - 1) / BM);
            wmma_gemm_kernel<<<g4, 256, 0, stream>>>(IMID, ew2, eb2, INTER, Me, D, F, 0);
        }
        ctx_gate_kernel<<<B, 256, 0, stream>>>(CTX, gw1, gb1, gw2, gb2, WG + e * B, D);
        fuse_acc_kernel<<<(B * T * D + 255) / 256, 256, 0, stream>>>(
            x, INTRA, INTER, WG + e * B, COEFF, e, ps, out, B, T, D);
    }
}
